// Message_50070728737146
// MI455X (gfx1250) — matrix operand order repaired and checked by reading, not compiled, not run on hardware
//
#include <hip/hip_runtime.h>

#define NEDGES 320000
#define NNODES 10000
#define NF 128
#define NRBF 20
#define RCUTF 5.0f
#define FEPS 1e-8f
#define PI_F 3.14159265358979323846f

#define NW   4            // waves per block (wave32)
#define EPW  16           // edges per wave (WMMA M)
#define EPB  (NW * EPW)   // 64 edges per block

typedef __attribute__((ext_vector_type(16))) _Float16 v16h;
typedef __attribute__((ext_vector_type(8)))  float    v8f;

// ============ fragment-major LDS layout ============
// One WMMA operand fragment = 32 lanes x 16 halves = 512 halves (1 KB),
// stored lane-contiguous so a lane's 16 halves are one aligned 32B block:
//   slot = (frag*32 + lane)*16 + pos
//
// B-matrix 32x16 (KxN), same K order as A: lane = N%16 + 16*((K>>3)&1), pos = K%8 + 8*(K>=16)
__device__ __forceinline__ int b_pack_idx(int k, int n) {
  return ((((k >> 3) & 1) ? 16 : 0) + n) * 16 + (k & 7) + 8 * (k >> 4);
}
// A-matrix 16x32 (MxK), ISA 7.12.2: lanes 0-15 hold K={0..7,16..23} at pos 0..15,
// lanes 16-31 hold K={8..15,24..31}; lane = M + 16*((K>>3)&1)
__device__ __forceinline__ int a_pack_idx(int m, int k) {
  int half = (k >> 3) & 1;
  int pos  = k - 8 * (k >> 4) - 8 * half;
  return (m + 16 * half) * 16 + pos;
}

__device__ __forceinline__ v16h load_frag(const _Float16* __restrict__ s,
                                          int frag, int lane) {
  // aligned 32B -> two ds_load_b128
  return *(const v16h*)(s + (frag * 32 + lane) * 16);
}

// fast reciprocal: single v_rcp_f32 (~1 ulp), no IEEE div fixup chain
__device__ __forceinline__ float fast_rcp(float x) {
  return __builtin_amdgcn_rcpf(x);
}

__global__ __launch_bounds__(NW * 32, 1)
void painn_message_kernel(const float* __restrict__ vj,
                          const float* __restrict__ sj,
                          const float* __restrict__ rij,
                          const float* __restrict__ W1,
                          const float* __restrict__ b1,
                          const float* __restrict__ W2,
                          const float* __restrict__ b2,
                          const float* __restrict__ Wrbf,
                          const float* __restrict__ brbf,
                          const int*   __restrict__ eij,
                          float* __restrict__ dvim,
                          float* __restrict__ dsim)
{
  // packed-fragment weight stores (frag index = nt*4 + ks for 128-K matrices)
  __shared__ __attribute__((aligned(32))) _Float16 W1p[32 * 512];   // 32 KB
  __shared__ __attribute__((aligned(32))) _Float16 W2p[32 * 512];   // 32 KB (one 128-col chunk)
  __shared__ __attribute__((aligned(32))) _Float16 Wrbfp[8 * 512];  //  8 KB (chunk, K 20->32 pad)
  __shared__ __attribute__((aligned(32))) _Float16 stageP[NW][4 * 512]; // sj then h, A-frags
  __shared__ __attribute__((aligned(32))) _Float16 rbfP[NW][512];       // RBF A-frag
  __shared__ float b1s[NF], b2s[3 * NF], brbfs[3 * NF];
  __shared__ float s1buf[NW][EPW * NF];   // 32 KB, s1 kept for the chunk-2 scatter
  __shared__ float fcuts[NW][EPW];
  __shared__ float rhats[NW][EPW][3];
  __shared__ int   dsts[NW][EPW];

  const int tid  = threadIdx.x;
  const int w    = tid >> 5;
  const int lane = tid & 31;
  const int e0   = (blockIdx.x * NW + w) * EPW;

  // ---- block-cooperative: W1 (f32->f16, pre-swizzled) + biases ----
  for (int idx = tid; idx < NF * NF; idx += NW * 32) {
    int k = idx >> 7, n = idx & 127;                 // W1[k][n], coalesced read
    int frag = (n >> 4) * 4 + (k >> 5);
    W1p[frag * 512 + b_pack_idx(k & 31, n & 15)] = (_Float16)W1[idx];
  }
  if (tid < NF) b1s[tid] = b1[tid];
  for (int i = tid; i < 3 * NF; i += NW * 32) { b2s[i] = b2[i]; brbfs[i] = brbf[i]; }

  // ---- per-wave: stage sj tile straight into A-fragment layout ----
  for (int t = lane; t < EPW * NF; t += 32) {
    int m = t >> 7, col = t & 127;
    stageP[w][(col >> 5) * 512 + a_pack_idx(m, col & 31)] =
        (_Float16)sj[(size_t)e0 * NF + t];
  }

  // ---- per-edge geometry + RBF (lanes 0..15, one edge each) ----
  if (lane < EPW) {
    int e = e0 + lane;
    float rx = rij[e * 3 + 0], ry = rij[e * 3 + 1], rz = rij[e * 3 + 2];
    float nrm = sqrtf(rx * rx + ry * ry + rz * rz);
    float inv = fast_rcp(nrm + FEPS);
    rhats[w][lane][0] = rx * inv;
    rhats[w][lane][1] = ry * inv;
    rhats[w][lane][2] = rz * inv;
    fcuts[w][lane] = (nrm > RCUTF) ? 0.0f : 0.5f * (__cosf(PI_F * nrm / RCUTF) + 1.0f);
    dsts[w][lane]  = eij[NEDGES + e];               // eij[1][e] (dst)
    float arg = PI_F * nrm * (1.0f / RCUTF);
#pragma unroll
    for (int t = 0; t < 32; ++t) {
      float v = (t < NRBF) ? __sinf((float)(t + 1) * arg) * inv : 0.0f;
      rbfP[w][a_pack_idx(lane, t)] = (_Float16)v;
    }
  }
  __syncthreads();

  // ---- A fragments in registers ----
  v16h sjA[4];
#pragma unroll
  for (int k = 0; k < 4; ++k) sjA[k] = load_frag(stageP[w], k, lane);
  v16h rbfA = load_frag(rbfP[w], 0, lane);

  const int hi8 = (lane & 16) ? 8 : 0;   // D-tile: M = r + hi8, N = lane&15
  const int nlo = lane & 15;

  // per-lane copies of the 8 edge scalars this lane's D rows touch
  float fc[8], rhx[8], rhy[8], rhz[8];
  int   dstn[8], erow[8];
#pragma unroll
  for (int r = 0; r < 8; ++r) {
    int M = r + hi8;
    fc[r]  = fcuts[w][M];
    rhx[r] = rhats[w][M][0];
    rhy[r] = rhats[w][M][1];
    rhz[r] = rhats[w][M][2];
    dstn[r] = dsts[w][M];
    erow[r] = e0 + M;
  }

  // ---- GEMM1: h = silu(sj @ W1 + b1), written back as packed A-frags ----
#pragma unroll 1
  for (int nt = 0; nt < 8; ++nt) {
    v8f acc = {};
#pragma unroll
    for (int k = 0; k < 4; ++k) {
      v16h bfr = load_frag(W1p, nt * 4 + k, lane);
      acc = __builtin_amdgcn_wmma_f32_16x16x32_f16(false, sjA[k], false, bfr,
                                                   (short)0, acc, false, false);
    }
    int col = nt * 16 + nlo;
    float bb = b1s[col];
#pragma unroll
    for (int r = 0; r < 8; ++r) {
      float x = acc[r] + bb;
      float h = x * fast_rcp(1.0f + __expf(-x));    // silu, v_rcp_f32 not IEEE div
      stageP[w][(col >> 5) * 512 + a_pack_idx(r + hi8, col & 31)] = (_Float16)h;
    }
  }
  __syncthreads();   // stage rewritten (same-wave ordered; barrier for safety)

  v16h hA[4];
#pragma unroll
  for (int k = 0; k < 4; ++k) hA[k] = load_frag(stageP[w], k, lane);

  // ---- three 128-column chunks: chunk0 -> s1, chunk1 -> s2, chunk2 -> s3 ----
#pragma unroll 1
  for (int c = 0; c < 3; ++c) {
    __syncthreads();   // all waves done with previous chunk's W2p/Wrbfp
    for (int idx = tid; idx < NF * NF; idx += NW * 32) {
      int k = idx >> 7, n = idx & 127;
      int frag = (n >> 4) * 4 + (k >> 5);
      W2p[frag * 512 + b_pack_idx(k & 31, n & 15)] =
          (_Float16)W2[k * (3 * NF) + c * NF + n];
    }
    for (int idx = tid; idx < 32 * NF; idx += NW * 32) {
      int k = idx >> 7, n = idx & 127;
      float v = (k < NRBF) ? Wrbf[k * (3 * NF) + c * NF + n] : 0.0f;
      Wrbfp[(n >> 4) * 512 + b_pack_idx(k, n & 15)] = (_Float16)v;
    }
    __syncthreads();

#pragma unroll 1
    for (int nt = 0; nt < 8; ++nt) {
      // T_RBF tile: one K=32 WMMA (K padded from 20 with zeros)
      v8f racc = {};
      {
        v16h bfr = load_frag(Wrbfp, nt, lane);
        racc = __builtin_amdgcn_wmma_f32_16x16x32_f16(false, rbfA, false, bfr,
                                                      (short)0, racc, false, false);
      }
      // phi tile: h @ W2 chunk
      v8f acc = {};
#pragma unroll
      for (int k = 0; k < 4; ++k) {
        v16h bfr = load_frag(W2p, nt * 4 + k, lane);
        acc = __builtin_amdgcn_wmma_f32_16x16x32_f16(false, hA[k], false, bfr,
                                                     (short)0, acc, false, false);
      }
      int f    = nt * 16 + nlo;   // feature within chunk
      int gcol = c * NF + f;      // feature within 384
      float bb2 = b2s[gcol], bbr = brbfs[gcol];
#pragma unroll
      for (int r = 0; r < 8; ++r) {
        float ws = (racc[r] + bbr) * fc[r];
        float p  = (acc[r] + bb2) * ws;            // phiW element
        if (c == 0) {
          s1buf[w][(r + hi8) * NF + f] = p;        // s1, reused at c==2
        } else if (c == 1) {
          // d_sim scatter: lanes 0-15 share an edge -> contiguous 64B per half-wave
          unsafeAtomicAdd(&dsim[(size_t)dstn[r] * NF + f], p);
        } else {
          float s1 = s1buf[w][(r + hi8) * NF + f];
          const float* vp = vj + ((size_t)erow[r] * NF + f) * 3;
          float mx = vp[0] * s1 + p * rhx[r];
          float my = vp[1] * s1 + p * rhy[r];
          float mz = vp[2] * s1 + p * rhz[r];
          float* op = dvim + ((size_t)dstn[r] * NF + f) * 3;
          unsafeAtomicAdd(op + 0, mx);
          unsafeAtomicAdd(op + 1, my);
          unsafeAtomicAdd(op + 2, mz);
        }
      }
    }
  }
}

extern "C" void kernel_launch(void* const* d_in, const int* in_sizes, int n_in,
                              void* d_out, int out_size, void* d_ws, size_t ws_size,
                              hipStream_t stream) {
  (void)in_sizes; (void)n_in; (void)d_ws; (void)ws_size;
  const float* vj   = (const float*)d_in[0];
  const float* sj   = (const float*)d_in[1];
  const float* rij  = (const float*)d_in[2];
  const float* W1   = (const float*)d_in[3];
  const float* b1   = (const float*)d_in[4];
  const float* W2   = (const float*)d_in[5];
  const float* b2   = (const float*)d_in[6];
  const float* Wrbf = (const float*)d_in[7];
  const float* brbf = (const float*)d_in[8];
  const int*   eij  = (const int*)d_in[9];

  float* dvim = (float*)d_out;                         // (N_NODES, NF, 3)
  float* dsim = dvim + (size_t)NNODES * NF * 3;        // (N_NODES, NF)

  // Outputs are accumulated with atomics: zero them first (capture-safe memset).
  hipMemsetAsync(d_out, 0, (size_t)out_size * sizeof(float), stream);

  dim3 grid(NEDGES / EPB);   // 320000 / 64 = 5000 full blocks, no tail
  dim3 block(NW * 32);
  painn_message_kernel<<<grid, block, 0, stream>>>(vj, sj, rij, W1, b1, W2, b2,
                                                   Wrbf, brbf, eij, dvim, dsim);
}
